// pointASNL_68650757259521
// MI455X (gfx1250) — matrix operand order repaired and checked by reading, not compiled, not run on hardware
//
#include <hip/hip_runtime.h>

#define DEV __device__ __forceinline__

typedef __attribute__((ext_vector_type(16))) _Float16 v16h;
typedef __attribute__((ext_vector_type(8)))  float    v8f;
typedef __attribute__((__vector_size__(16))) int      v4i_b128;

constexpr int   B_  = 8;
constexpr int   N_  = 4096;
constexpr int   S_  = 1024;
constexpr float BNC = 0.99999500003749973f;   // 1/sqrt(1+1e-5)

// ---------------- gfx1250 async global->LDS path (guarded) ----------------

#if defined(__AMDGCN__) && __has_builtin(__builtin_amdgcn_global_load_async_to_lds_b128) && __has_builtin(__builtin_amdgcn_s_wait_asynccnt)
#define HAS_ASYNC 1
#else
#define HAS_ASYNC 0
#endif

DEV void async_cp_b128(const _Float16* g, _Float16* l) {
#if HAS_ASYNC
  __builtin_amdgcn_global_load_async_to_lds_b128(
      (__attribute__((address_space(1))) v4i_b128*)g,
      (__attribute__((address_space(3))) v4i_b128*)l, 0, 0);
#else
  *(int4*)(void*)l = *(const int4*)(const void*)g;
#endif
}

DEV void async_wait_le6() {
#if HAS_ASYNC
  __builtin_amdgcn_s_wait_asynccnt(6);
#endif
}

// ---------------- WMMA fragment helpers (wave32, 16x16x32 f16) ----------------

DEV v8f zero8() {
  v8f z;
#pragma unroll
  for (int r = 0; r < 8; ++r) z[r] = 0.f;
  return z;
}

DEV v8f wmma32(v16h a, v16h b, v8f c) {
  return __builtin_amdgcn_wmma_f32_16x16x32_f16(false, a, false, b, (short)0, c,
                                                false, false);
}

// A: 16x32 f16, row-major with leading dim lda; ptr at tile origin.
DEV v16h frag_a(const _Float16* A, int lda) {
  int lane = threadIdx.x & 31;
  int hf = lane >> 4, m = lane & 15;
  v16h a;
#pragma unroll
  for (int r = 0; r < 8; ++r) {
    int kb = (r < 4) ? (hf * 8 + 2 * r) : (16 + hf * 8 + 2 * (r - 4));
    a[2 * r]     = A[m * lda + kb];
    a[2 * r + 1] = A[m * lda + kb + 1];
  }
  return a;
}

// A logically 16x32, but source is stored transposed: A[m][k] = src[k*lda + m]
DEV v16h frag_aT(const _Float16* A, int lda) {
  int lane = threadIdx.x & 31;
  int hf = lane >> 4, m = lane & 15;
  v16h a;
#pragma unroll
  for (int r = 0; r < 8; ++r) {
    int kb = (r < 4) ? (hf * 8 + 2 * r) : (16 + hf * 8 + 2 * (r - 4));
    a[2 * r]     = A[kb * lda + m];
    a[2 * r + 1] = A[(kb + 1) * lda + m];
  }
  return a;
}

// A built from an f32 source (converted on the fly)
DEV v16h frag_a_f32(const float* A, int lda) {
  int lane = threadIdx.x & 31;
  int hf = lane >> 4, m = lane & 15;
  v16h a;
#pragma unroll
  for (int r = 0; r < 8; ++r) {
    int kb = (r < 4) ? (hf * 8 + 2 * r) : (16 + hf * 8 + 2 * (r - 4));
    a[2 * r]     = (_Float16)A[m * lda + kb];
    a[2 * r + 1] = (_Float16)A[m * lda + kb + 1];
  }
  return a;
}

// B: 32x16 f16, row-major (K x N) with leading dim ldb; ptr at tile origin.
DEV v16h frag_b(const _Float16* Bp, int ldb) {
  int lane = threadIdx.x & 31;
  int hf = lane >> 4, n = lane & 15;
  v16h b;
#pragma unroll
  for (int r = 0; r < 8; ++r) {
    int kb = (r < 4) ? (hf * 8 + 2 * r) : (16 + hf * 8 + 2 * (r - 4));
    b[2 * r]     = Bp[kb * ldb + n];
    b[2 * r + 1] = Bp[(kb + 1) * ldb + n];
  }
  return b;
}

// B logically 32x16, source stored transposed: B[k][n] = src[n*ld + k]
DEV v16h frag_bT(const _Float16* Bp, int ld) {
  int lane = threadIdx.x & 31;
  int hf = lane >> 4, n = lane & 15;
  v16h b;
#pragma unroll
  for (int r = 0; r < 8; ++r) {
    int kb = (r < 4) ? (hf * 8 + 2 * r) : (16 + hf * 8 + 2 * (r - 4));
    b[2 * r]     = Bp[n * ld + kb];
    b[2 * r + 1] = Bp[n * ld + kb + 1];
  }
  return b;
}

// D/C layout: VGPR r -> row (half*8 + r), col = lane&15
DEV void store_f32(float* D, int ldd, v8f acc) {
  int lane = threadIdx.x & 31;
  int hf = lane >> 4, n = lane & 15;
#pragma unroll
  for (int r = 0; r < 8; ++r) D[(hf * 8 + r) * ldd + n] = acc[r];
}

DEV void store_f16(_Float16* D, int ldd, v8f acc) {
  int lane = threadIdx.x & 31;
  int hf = lane >> 4, n = lane & 15;
#pragma unroll
  for (int r = 0; r < 8; ++r) D[(hf * 8 + r) * ldd + n] = (_Float16)acc[r];
}

// 32-row two-M-tile GEMM: F32 out = A(32 x Ktiles*32) @ W(Ktiles*32 x Ntiles*16)
DEV void mlp_gemm(const _Float16* Ain, int lda, const _Float16* W,
                  int Ktiles, int Ntiles, float* Fout) {
  int ncols = Ntiles * 16;
  for (int mt = 0; mt < 2; ++mt)
    for (int nt = 0; nt < Ntiles; ++nt) {
      v8f acc = zero8();
      for (int kt = 0; kt < Ktiles; ++kt) {
        v16h a = frag_a(Ain + mt * 16 * lda + kt * 32, lda);
        v16h w = frag_b(W + kt * 32 * ncols + nt * 16, ncols);
        acc = wmma32(a, w, acc);
      }
      store_f32(Fout + mt * 16 * ncols + nt * 16, ncols, acc);
    }
}

DEV void bias_relu_f16(const float* F, const float* bias, int nchan, int ld,
                       _Float16* Aout, int tid) {
  for (int i = tid; i < 32 * ld; i += 32) {
    int c = i % ld;
    float bv = (c < nchan) ? bias[c] : 0.f;
    Aout[i] = (_Float16)fmaxf((F[i] + bv) * BNC, 0.f);
  }
}

// ---------------- small prep kernels ----------------

__global__ void k_newxyz(const float* xyz, const int* cidx, float* new_xyz) {
  int i = blockIdx.x * blockDim.x + threadIdx.x;   // over B*S
  if (i >= B_ * S_) return;
  int b = i >> 10;
  int n = cidx[i];
  for (int c = 0; c < 3; ++c)
    new_xyz[i * 3 + c] = xyz[((size_t)b * 3 + c) * N_ + n];
}

__global__ void k_pad(const float* src, _Float16* dst, int K, int N, int Kp, int Np) {
  int i = blockIdx.x * blockDim.x + threadIdx.x;
  if (i >= Kp * Np) return;
  int k = i / Np, n = i % Np;
  dst[i] = (_Float16)((k < K && n < N) ? src[k * N + n] : 0.f);
}

// WpcR[(c*32+j)*128 + o] = Wpc[(o*32+j)*128 + c]
__global__ void k_wpcr(const float* Wpc, _Float16* WpcR) {
  int i = blockIdx.x * blockDim.x + threadIdx.x;
  if (i >= 4096 * 128) return;
  int row = i / 128, o = i % 128;
  int c = row >> 5, j = row & 31;
  WpcR[i] = (_Float16)Wpc[((size_t)o * 32 + j) * 128 + c];
}

// ---------------- top-K neighbor search + gp tile build ----------------

__global__ void __launch_bounds__(256)
k_topk(const float* xyz, const float* new_xyz, const float* points, _Float16* gp) {
  int bs = blockIdx.x;              // 0..8191
  int b = bs >> 10;
  int tid = threadIdx.x;
  __shared__ float dist[N_];
  __shared__ float rv[256];
  __shared__ int   ri[256];
  __shared__ int   sel[32];

  float cx = new_xyz[bs * 3 + 0];
  float cy = new_xyz[bs * 3 + 1];
  float cz = new_xyz[bs * 3 + 2];
  for (int n = tid; n < N_; n += 256) {
    float dx = xyz[((size_t)b * 3 + 0) * N_ + n] - cx;
    float dy = xyz[((size_t)b * 3 + 1) * N_ + n] - cy;
    float dz = xyz[((size_t)b * 3 + 2) * N_ + n] - cz;
    dist[n] = dx * dx + dy * dy + dz * dz;
  }
  __syncthreads();

  for (int kk = 0; kk < 32; ++kk) {
    float bv = 1e30f;
    int bi = N_;
    for (int n = tid; n < N_; n += 256) {
      float dv = dist[n];
      if (dv < bv || (dv == bv && n < bi)) { bv = dv; bi = n; }
    }
    rv[tid] = bv;
    ri[tid] = bi;
    __syncthreads();
    for (int off = 128; off > 0; off >>= 1) {
      if (tid < off) {
        float ov = rv[tid + off]; int oi = ri[tid + off];
        if (ov < rv[tid] || (ov == rv[tid] && oi < ri[tid])) { rv[tid] = ov; ri[tid] = oi; }
      }
      __syncthreads();
    }
    if (tid == 0) { sel[kk] = ri[0]; dist[ri[0]] = 1e30f; }
    __syncthreads();
  }

  // gp[bs][k][c]: c<3 -> gxyz - new_xyz, c in [3,67) -> gfeat, rest 0 (pad to 96)
  for (int i = tid; i < 32 * 96; i += 256) {
    int k = i / 96, c = i % 96;
    int n = sel[k];
    float val;
    if (c < 3)       val = xyz[((size_t)b * 3 + c) * N_ + n] - new_xyz[bs * 3 + c];
    else if (c < 67) val = points[((size_t)b * 64 + (c - 3)) * N_ + n];
    else             val = 0.f;
    gp[(size_t)bs * 3072 + i] = (_Float16)val;
  }
}

// ---------------- fused per-neighborhood pipeline (1 wave per (b,s)) ----------------

__global__ void __launch_bounds__(32)
k_local(const float* new_xyz, const _Float16* gp,
        const _Float16* Wq_h, const _Float16* Wkv_h,
        const _Float16* Wm0_h, const float* bm0,
        const _Float16* Wm1_h, const float* bm1,
        const _Float16* Wm2_h, const float* bm2,
        const _Float16* Wl0_h, const float* bl0,
        const _Float16* Wl1_h, const float* bl1,
        const _Float16* Wl2_h, const float* bl2,
        const float* Wsk, const float* Wwn, const float* bwn,
        float* out_xyz, float* new_feat, float* skip, _Float16* lp2v) {
  int bs = blockIdx.x;
  int b = bs >> 10, s = bs & 1023;
  int tid = threadIdx.x;

  __shared__ _Float16 sGP[32 * 96];
  __shared__ _Float16 sA [32 * 128];
  __shared__ _Float16 sK [32 * 64];
  __shared__ _Float16 sV [32 * 64];
  __shared__ _Float16 sQ [32 * 64];
  __shared__ float    sF [32 * 128];
  __shared__ float    sMisc[256];

  for (int i = tid; i < 32 * 96; i += 32) sGP[i] = gp[(size_t)bs * 3072 + i];
  __syncthreads();

  // q = gp @ Wq (32x96 @ 96x64) -> sQ (f16)
  for (int mt = 0; mt < 2; ++mt)
    for (int nt = 0; nt < 4; ++nt) {
      v8f acc = zero8();
      for (int kt = 0; kt < 3; ++kt) {
        v16h a = frag_a(sGP + mt * 16 * 96 + kt * 32, 96);
        v16h w = frag_b(Wq_h + kt * 32 * 64 + nt * 16, 64);
        acc = wmma32(a, w, acc);
      }
      store_f16(sQ + mt * 16 * 64 + nt * 16, 64, acc);
    }
  // kv = gp @ Wkv (32x96 @ 96x128) -> sK | sV
  for (int mt = 0; mt < 2; ++mt)
    for (int nt = 0; nt < 8; ++nt) {
      v8f acc = zero8();
      for (int kt = 0; kt < 3; ++kt) {
        v16h a = frag_a(sGP + mt * 16 * 96 + kt * 32, 96);
        v16h w = frag_b(Wkv_h + kt * 32 * 128 + nt * 16, 128);
        acc = wmma32(a, w, acc);
      }
      if (nt < 4) store_f16(sK + mt * 16 * 64 + nt * 16, 64, acc);
      else        store_f16(sV + mt * 16 * 64 + (nt - 4) * 16, 64, acc);
    }
  __syncthreads();

  // att = softmax(q @ k^T / 8)
  for (int mt = 0; mt < 2; ++mt)
    for (int nt = 0; nt < 2; ++nt) {
      v8f acc = zero8();
      for (int kt = 0; kt < 2; ++kt) {
        v16h a = frag_a(sQ + mt * 16 * 64 + kt * 32, 64);
        v16h w = frag_bT(sK + nt * 16 * 64 + kt * 32, 64);
        acc = wmma32(a, w, acc);
      }
#pragma unroll
      for (int r = 0; r < 8; ++r) acc[r] *= 0.125f;
      store_f32(sF + mt * 16 * 32 + nt * 16, 32, acc);
    }
  __syncthreads();
  {
    float mx = -1e30f;
    for (int j = 0; j < 32; ++j) mx = fmaxf(mx, sF[tid * 32 + j]);
    float sum = 0.f;
    for (int j = 0; j < 32; ++j) sum += __expf(sF[tid * 32 + j] - mx);
    float inv = 1.f / sum;
    for (int j = 0; j < 32; ++j)
      sA[tid * 32 + j] = (_Float16)(__expf(sF[tid * 32 + j] - mx) * inv);
  }
  __syncthreads();

  // h = att @ v (32x32 @ 32x64) -> sF (f32, ld 64)
  for (int mt = 0; mt < 2; ++mt)
    for (int nt = 0; nt < 4; ++nt) {
      v8f acc = zero8();
      v16h a = frag_a(sA + mt * 16 * 32, 32);
      v16h w = frag_b(sV + nt * 16, 64);
      acc = wmma32(a, w, acc);
      store_f32(sF + mt * 16 * 64 + nt * 16, 64, acc);
    }
  __syncthreads();
  for (int i = tid; i < 32 * 64; i += 32) sA[i] = (_Float16)sF[i];
  __syncthreads();

  // MLP: 64->64->64->65(pad80), relu((x@W+b)*BN)
  mlp_gemm(sA, 64, Wm0_h, 2, 4, sF); __syncthreads();
  bias_relu_f16(sF, bm0, 64, 64, sA, tid); __syncthreads();
  mlp_gemm(sA, 64, Wm1_h, 2, 4, sF); __syncthreads();
  bias_relu_f16(sF, bm1, 64, 64, sA, tid); __syncthreads();
  mlp_gemm(sA, 64, Wm2_h, 2, 5, sF); __syncthreads();
  for (int i = tid; i < 32 * 80; i += 32) {
    int c = i % 80;
    float bv = (c < 65) ? bm2[c] : 0.f;
    sF[i] = fmaxf((sF[i] + bv) * BNC, 0.f);
  }
  __syncthreads();

  // w_ = softmax over neighbors (rows) per channel
  for (int c = tid; c < 65; c += 32) {
    float mx = -1e30f;
    for (int k = 0; k < 32; ++k) mx = fmaxf(mx, sF[k * 80 + c]);
    float sum = 0.f;
    for (int k = 0; k < 32; ++k) sum += __expf(sF[k * 80 + c] - mx);
    float inv = 1.f / sum;
    for (int k = 0; k < 32; ++k) sF[k * 80 + c] = __expf(sF[k * 80 + c] - mx) * inv;
  }
  __syncthreads();

  // new_xyz2 and new_feat
  if (tid < 3) {
    float acc = 0.f;
    for (int k = 0; k < 32; ++k)
      acc += sF[k * 80] * ((float)sGP[k * 96 + tid] + new_xyz[bs * 3 + tid]);
    sMisc[tid] = acc;
    out_xyz[((size_t)b * 3 + tid) * S_ + s] = acc;
  }
  for (int c = tid; c < 64; c += 32) {
    float acc = 0.f;
    for (int k = 0; k < 32; ++k)
      acc += sF[k * 80 + 1 + c] * (float)sGP[k * 96 + 3 + c];
    new_feat[(size_t)bs * 64 + c] = acc;
  }
  __syncthreads();

  // np_cat = [gxyz-new_xyz2 | gxyz | gfeat] padded to 96, f16 in sA (ld 96)
  for (int i = tid; i < 32 * 96; i += 32) {
    int k = i / 96, c = i % 96;
    float val;
    if (c < 3)       val = (float)sGP[k * 96 + c] + new_xyz[bs * 3 + c] - sMisc[c];
    else if (c < 6)  val = (float)sGP[k * 96 + c - 3] + new_xyz[bs * 3 + c - 3];
    else if (c < 70) val = (float)sGP[k * 96 + c - 3];
    else             val = 0.f;
    sA[i] = (_Float16)val;
  }
  __syncthreads();

  // skip = relu(max_k(np_cat) @ Wsk * BN)
  for (int c = tid; c < 70; c += 32) {
    float mx = -1e30f;
    for (int k = 0; k < 32; ++k) mx = fmaxf(mx, (float)sA[k * 96 + c]);
    sMisc[16 + c] = mx;
  }
  __syncthreads();
  for (int o = tid; o < 128; o += 32) {
    float acc = 0.f;
    for (int c = 0; c < 70; ++c) acc += sMisc[16 + c] * Wsk[c * 128 + o];
    skip[(size_t)bs * 128 + o] = fmaxf(acc * BNC, 0.f);
  }

  // lp MLP: 96->64->64->128
  mlp_gemm(sA, 96, Wl0_h, 3, 4, sF); __syncthreads();
  bias_relu_f16(sF, bl0, 64, 64, sK, tid); __syncthreads();
  mlp_gemm(sK, 64, Wl1_h, 2, 4, sF); __syncthreads();
  bias_relu_f16(sF, bl1, 64, 64, sV, tid); __syncthreads();
  mlp_gemm(sV, 64, Wl2_h, 2, 8, sF); __syncthreads();
  bias_relu_f16(sF, bl2, 128, 128, sA, tid); __syncthreads();

  // wn = relu(((gxyz-new_xyz2) @ Wwn + bwn) * BN), 32x32 -> sQ
  {
    int j = tid;
    for (int k = 0; k < 32; ++k) {
      float acc = bwn[j];
      for (int c = 0; c < 3; ++c)
        acc += ((float)sGP[k * 96 + c] + new_xyz[bs * 3 + c] - sMisc[c]) * Wwn[c * 32 + j];
      sQ[k * 32 + j] = (_Float16)fmaxf(acc * BNC, 0.f);
    }
  }
  __syncthreads();

  // lp2[c][j] = sum_k lp[k][c]*wn[k][j]  (128x32 via lp^T @ wn), store flattened f16
  for (int mt = 0; mt < 8; ++mt)
    for (int nt = 0; nt < 2; ++nt) {
      v8f acc = zero8();
      v16h a = frag_aT(sA + mt * 16, 128);
      v16h w = frag_b(sQ + nt * 16, 32);
      acc = wmma32(a, w, acc);
      store_f16(lp2v + (size_t)bs * 4096 + (mt * 16) * 32 + nt * 16, 32, acc);
    }
}

// ---------------- projections for non-local attention ----------------

// C_f16[M x N] = A_f32[M x K] @ B_f16[K x N]; one wave per 16 rows
__global__ void k_gemm(const float* A, const _Float16* Bm, _Float16* C,
                       int M, int N, int K) {
  int w = (blockIdx.x * blockDim.x + threadIdx.x) >> 5;
  if (w >= M / 16) return;
  const float* Arow = A + (size_t)w * 16 * K;
  for (int nt = 0; nt < N / 16; ++nt) {
    v8f acc = zero8();
    for (int kt = 0; kt < K / 32; ++kt) {
      v16h a = frag_a_f32(Arow + kt * 32, K);
      v16h wb = frag_b(Bm + (size_t)kt * 32 * N + nt * 16, N);
      acc = wmma32(a, wb, acc);
    }
    store_f16(C + (size_t)w * 16 * N + nt * 16, N, acc);
  }
}

// nlk|nlv = pts @ Wnlkv ; pts read transposed from points(B,D,N)
__global__ void k_nlkv(const float* points, const _Float16* Wnlkv_h,
                       _Float16* nlk, _Float16* nlv) {
  int w = (blockIdx.x * blockDim.x + threadIdx.x) >> 5;
  if (w >= 2048) return;
  int row0 = w * 16;
  int b = row0 >> 12;
  int n0 = row0 & 4095;
  int lane = threadIdx.x & 31;
  int hf = lane >> 4, m = lane & 15;
  for (int nt = 0; nt < 8; ++nt) {
    v8f acc = zero8();
    for (int kt = 0; kt < 2; ++kt) {
      v16h a;
#pragma unroll
      for (int r = 0; r < 8; ++r) {
        int kb = (r < 4) ? (hf * 8 + 2 * r) : (16 + hf * 8 + 2 * (r - 4));
        int kk = kt * 32 + kb;
        a[2 * r]     = (_Float16)points[((size_t)b * 64 + kk) * N_ + n0 + m];
        a[2 * r + 1] = (_Float16)points[((size_t)b * 64 + kk + 1) * N_ + n0 + m];
      }
      v16h wb = frag_b(Wnlkv_h + kt * 32 * 128 + nt * 16, 128);
      acc = wmma32(a, wb, acc);
    }
    if (nt < 4) store_f16(nlk + (size_t)row0 * 64 + nt * 16, 64, acc);
    else        store_f16(nlv + (size_t)row0 * 64 + (nt - 4) * 16, 64, acc);
  }
}

// ---------------- flash-style non-local attention + fused @Wnl ----------------

__global__ void __launch_bounds__(32)
k_flash(const _Float16* nlq, const _Float16* nlk, const _Float16* nlv,
        const _Float16* Wnl_h, float* nl) {
  int w = blockIdx.x;             // 0..511, 16 query rows each
  int row0 = w * 16;
  int b = row0 >> 10;
  int tid = threadIdx.x;
  int hf = tid >> 4, ncol = tid & 15;

  __shared__ _Float16 sQ[16 * 64];
  __shared__ float    sS[16 * 32];
  __shared__ _Float16 sP[16 * 32];
  __shared__ float    sAl[16];
  __shared__ float    sL[16];
  __shared__ _Float16 sO[16 * 64];

  for (int i = tid; i < 16 * 64; i += 32) sQ[i] = nlq[(size_t)row0 * 64 + i];
  __syncthreads();

  v8f o0 = zero8(), o1 = zero8(), o2 = zero8(), o3 = zero8();
  float mrun = -1e30f, lrun = 0.f;      // live in lanes 0..15
  const _Float16* Kb = nlk + (size_t)b * N_ * 64;
  const _Float16* Vb = nlv + (size_t)b * N_ * 64;

  for (int j0 = 0; j0 < N_; j0 += 32) {
    // prefetch next chunk of K/V while this chunk computes
    int jn = (j0 + 32) & (N_ - 1);
    __builtin_prefetch(Kb + (size_t)jn * 64 + tid * 64, 0, 1);
    __builtin_prefetch(Vb + (size_t)jn * 64 + tid * 64, 0, 1);

    for (int nt = 0; nt < 2; ++nt) {
      v8f acc = zero8();
      for (int kt = 0; kt < 2; ++kt) {
        v16h a = frag_a(sQ + kt * 32, 64);
        v16h wb = frag_bT(Kb + (size_t)(j0 + nt * 16) * 64 + kt * 32, 64);
        acc = wmma32(a, wb, acc);
      }
#pragma unroll
      for (int r = 0; r < 8; ++r) acc[r] *= 0.125f;
      store_f32(sS + nt * 16, 32, acc);
    }
    __syncthreads();
    if (tid < 16) {
      float mx = mrun;
      for (int j = 0; j < 32; ++j) mx = fmaxf(mx, sS[tid * 32 + j]);
      float alpha = __expf(mrun - mx);
      float ls = 0.f;
      for (int j = 0; j < 32; ++j) {
        float e = __expf(sS[tid * 32 + j] - mx);
        sP[tid * 32 + j] = (_Float16)e;
        ls += e;
      }
      lrun = lrun * alpha + ls;
      mrun = mx;
      sAl[tid] = alpha;
    }
    __syncthreads();
#pragma unroll
    for (int r = 0; r < 8; ++r) {
      float al = sAl[hf * 8 + r];
      o0[r] *= al; o1[r] *= al; o2[r] *= al; o3[r] *= al;
    }
    v16h p = frag_a(sP, 32);
    o0 = wmma32(p, frag_b(Vb + (size_t)j0 * 64 + 0, 64), o0);
    o1 = wmma32(p, frag_b(Vb + (size_t)j0 * 64 + 16, 64), o1);
    o2 = wmma32(p, frag_b(Vb + (size_t)j0 * 64 + 32, 64), o2);
    o3 = wmma32(p, frag_b(Vb + (size_t)j0 * 64 + 48, 64), o3);
  }

  if (tid < 16) sL[tid] = 1.f / lrun;
  __syncthreads();
#pragma unroll
  for (int r = 0; r < 8; ++r) {
    float inv = sL[hf * 8 + r];
    o0[r] *= inv; o1[r] *= inv; o2[r] *= inv; o3[r] *= inv;
  }
  store_f16(sO + 0, 64, o0);
  store_f16(sO + 16, 64, o1);
  store_f16(sO + 32, 64, o2);
  store_f16(sO + 48, 64, o3);
  __syncthreads();

  // nl = relu((O @ Wnl) * BN)   16x64 @ 64x128
  for (int nt = 0; nt < 8; ++nt) {
    v8f acc = zero8();
    for (int kt = 0; kt < 2; ++kt) {
      v16h a = frag_a(sO + kt * 32, 64);
      v16h wb = frag_b(Wnl_h + kt * 32 * 128 + nt * 16, 128);
      acc = wmma32(a, wb, acc);
    }
#pragma unroll
    for (int r = 0; r < 8; ++r)
      nl[(size_t)(row0 + hf * 8 + r) * 128 + nt * 16 + ncol] =
          fmaxf(acc[r] * BNC, 0.f);
  }
}

// ---------------- loc = relu((lp2v @ WpcR + bpc) * BN) : 8192 x 4096 x 128 ----
// LDS-tiled, double-buffered via gfx1250 async global->LDS B128 transfers.
// Block = 4 waves; B panel (32x128 f16) shared by block, per-wave A tile 16x32.

DEV void kloc_stage(const _Float16* WpcR, const _Float16* lp2v, int row0, int kt,
                    _Float16* sBbuf, _Float16* sAbuf, int t) {
  // B panel: rows kt*32..kt*32+31 of WpcR are contiguous (8 KB) -> 512 x b128
  const _Float16* bsrc = WpcR + (size_t)kt * 32 * 128;
#pragma unroll
  for (int i = 0; i < 4; ++i) {
    int j = t + i * 128;
    async_cp_b128(bsrc + j * 8, sBbuf + j * 8);
  }
  // A tile: 16 rows x 32 halves (64B per row) -> 64 x b128, 2 per lane
  int lane = t & 31;
  const _Float16* asrc = lp2v + (size_t)row0 * 4096 + kt * 32;
#pragma unroll
  for (int i = 0; i < 2; ++i) {
    int j = lane + i * 32;
    int row = j >> 2, seg = j & 3;
    async_cp_b128(asrc + (size_t)row * 4096 + seg * 8, sAbuf + row * 32 + seg * 8);
  }
}

__global__ void __launch_bounds__(128)
k_loc(const _Float16* lp2v, const _Float16* WpcR, const float* bpc, float* loc) {
  int t = threadIdx.x;
  int wv = t >> 5;
  int row0 = (blockIdx.x * 4 + wv) * 16;
  int lane = t & 31;
  int hf = lane >> 4, n = lane & 15;

  __shared__ alignas(16) _Float16 sB[2][32 * 128];
  __shared__ alignas(16) _Float16 sA[4][2][16 * 32];

  v8f acc[8];
  for (int nt = 0; nt < 8; ++nt) acc[nt] = zero8();

  kloc_stage(WpcR, lp2v, row0, 0, sB[0], sA[wv][0], t);
  for (int kt = 0; kt < 128; ++kt) {
    int nkt = (kt + 1) & 127;                 // wrap: harmless re-prefetch at end
    kloc_stage(WpcR, lp2v, row0, nkt, sB[(kt + 1) & 1], sA[wv][(kt + 1) & 1], t);
    async_wait_le6();                          // previous batch (6/lane) complete
    __syncthreads();
    v16h a = frag_a(sA[wv][kt & 1], 32);
#pragma unroll
    for (int nt = 0; nt < 8; ++nt) {
      v16h wb = frag_b(&sB[kt & 1][nt * 16], 128);
      acc[nt] = wmma32(a, wb, acc[nt]);
    }
    __syncthreads();
  }

  for (int nt = 0; nt < 8; ++nt) {
#pragma unroll
    for (int r = 0; r < 8; ++r) {
      float v = fmaxf((acc[nt][r] + bpc[nt * 16 + n]) * BNC, 0.f);
      loc[(size_t)(row0 + hf * 8 + r) * 128 + nt * 16 + n] = v;
    }
  }
}

// ---------------- out = relu((loc + nl + skip) @ Wff * BN), transposed store ----------------

__global__ void k_final(const float* loc, const float* nl, const float* skip,
                        const _Float16* Wff_h, float* out) {
  int w = (blockIdx.x * blockDim.x + threadIdx.x) >> 5;
  if (w >= 512) return;
  int row0 = w * 16;
  int lane = threadIdx.x & 31;
  int hf = lane >> 4, m = lane & 15, n = lane & 15;
  v16h afr[4];
  for (int kt = 0; kt < 4; ++kt) {
#pragma unroll
    for (int r = 0; r < 8; ++r) {
      int kb = (r < 4) ? (hf * 8 + 2 * r) : (16 + hf * 8 + 2 * (r - 4));
      size_t idx = (size_t)(row0 + m) * 128 + kt * 32 + kb;
      afr[kt][2 * r]     = (_Float16)(loc[idx] + nl[idx] + skip[idx]);
      afr[kt][2 * r + 1] = (_Float16)(loc[idx + 1] + nl[idx + 1] + skip[idx + 1]);
    }
  }
  for (int nt = 0; nt < 8; ++nt) {
    v8f acc = zero8();
    for (int kt = 0; kt < 4; ++kt) {
      v16h wb = frag_b(Wff_h + kt * 32 * 128 + nt * 16, 128);
      acc = wmma32(afr[kt], wb, acc);
    }
#pragma unroll
    for (int r = 0; r < 8; ++r) {
      int row = row0 + hf * 8 + r;        // global row = b*1024 + s
      int b = row >> 10, s = row & 1023;
      int col = nt * 16 + n;
      out[((size_t)b * 128 + col) * S_ + s] = fmaxf(acc[r] * BNC, 0.f);
    }
  }
}

// ---------------- host orchestration ----------------

extern "C" void kernel_launch(void* const* d_in, const int* in_sizes, int n_in,
                              void* d_out, int out_size, void* d_ws, size_t ws_size,
                              hipStream_t stream) {
  (void)in_sizes; (void)n_in; (void)out_size; (void)ws_size;
  const float* xyz    = (const float*)d_in[0];
  const float* points = (const float*)d_in[1];
  const int*   cidx   = (const int*)d_in[2];
  const float* Wq   = (const float*)d_in[3];
  const float* Wkv  = (const float*)d_in[4];
  const float* Wm0  = (const float*)d_in[5];
  const float* bm0  = (const float*)d_in[6];
  const float* Wm1  = (const float*)d_in[7];
  const float* bm1  = (const float*)d_in[8];
  const float* Wm2  = (const float*)d_in[9];
  const float* bm2  = (const float*)d_in[10];
  const float* Wl0  = (const float*)d_in[11];
  const float* bl0  = (const float*)d_in[12];
  const float* Wl1  = (const float*)d_in[13];
  const float* bl1  = (const float*)d_in[14];
  const float* Wl2  = (const float*)d_in[15];
  const float* bl2  = (const float*)d_in[16];
  const float* Wwn  = (const float*)d_in[17];
  const float* bwn  = (const float*)d_in[18];
  const float* Wpc  = (const float*)d_in[19];
  const float* bpc  = (const float*)d_in[20];
  const float* Wsk  = (const float*)d_in[21];
  const float* Wnlq = (const float*)d_in[22];
  const float* Wnlkv= (const float*)d_in[23];
  const float* Wnl  = (const float*)d_in[24];
  const float* Wff  = (const float*)d_in[25];

  char* ws = (char*)d_ws;
  size_t off = 0;
  auto take = [&](size_t bytes) -> char* {
    char* p = ws + off;
    off = (off + bytes + 255) & ~(size_t)255;
    return p;
  };

  float*    new_xyz  = (float*)take((size_t)8192 * 3 * 4);
  _Float16* gp       = (_Float16*)take((size_t)8192 * 32 * 96 * 2);
  _Float16* Wq_h     = (_Float16*)take(96 * 64 * 2);
  _Float16* Wkv_h    = (_Float16*)take(96 * 128 * 2);
  _Float16* Wm0_h    = (_Float16*)take(64 * 64 * 2);
  _Float16* Wm1_h    = (_Float16*)take(64 * 64 * 2);
  _Float16* Wm2_h    = (_Float16*)take(64 * 80 * 2);
  _Float16* Wl0_h    = (_Float16*)take(96 * 64 * 2);
  _Float16* Wl1_h    = (_Float16*)take(64 * 64 * 2);
  _Float16* Wl2_h    = (_Float16*)take(64 * 128 * 2);
  _Float16* WpcR     = (_Float16*)take((size_t)4096 * 128 * 2);
  _Float16* Wnlq_h   = (_Float16*)take(64 * 64 * 2);
  _Float16* Wnlkv_h  = (_Float16*)take(64 * 128 * 2);
  _Float16* Wnl_h    = (_Float16*)take(64 * 128 * 2);
  _Float16* Wff_h    = (_Float16*)take(128 * 128 * 2);
  float*    new_feat = (float*)take((size_t)8192 * 64 * 4);
  float*    skip     = (float*)take((size_t)8192 * 128 * 4);
  _Float16* lp2v     = (_Float16*)take((size_t)8192 * 4096 * 2);
  _Float16* nlq_h    = (_Float16*)take((size_t)8192 * 64 * 2);
  _Float16* nlk_h    = (_Float16*)take((size_t)8 * 4096 * 64 * 2);
  _Float16* nlv_h    = (_Float16*)take((size_t)8 * 4096 * 64 * 2);
  float*    nlbuf    = (float*)take((size_t)8192 * 128 * 4);
  float*    locbuf   = (float*)take((size_t)8192 * 128 * 4);

  auto pad = [&](const float* src, _Float16* dst, int K, int N, int Kp, int Np) {
    int tot = Kp * Np;
    k_pad<<<(tot + 255) / 256, 256, 0, stream>>>(src, dst, K, N, Kp, Np);
  };

  k_newxyz<<<32, 256, 0, stream>>>(xyz, cidx, new_xyz);

  pad(Wq, Wq_h, 67, 64, 96, 64);
  pad(Wkv, Wkv_h, 67, 128, 96, 128);
  pad(Wm0, Wm0_h, 64, 64, 64, 64);
  pad(Wm1, Wm1_h, 64, 64, 64, 64);
  pad(Wm2, Wm2_h, 64, 65, 64, 80);
  pad(Wl0, Wl0_h, 70, 64, 96, 64);
  pad(Wl1, Wl1_h, 64, 64, 64, 64);
  pad(Wl2, Wl2_h, 64, 128, 64, 128);
  pad(Wnlq, Wnlq_h, 64, 64, 64, 64);
  pad(Wnlkv, Wnlkv_h, 64, 128, 64, 128);
  pad(Wnl, Wnl_h, 64, 128, 64, 128);
  pad(Wff, Wff_h, 128, 128, 128, 128);
  k_wpcr<<<2048, 256, 0, stream>>>(Wpc, WpcR);

  k_topk<<<8192, 256, 0, stream>>>(xyz, new_xyz, points, gp);

  k_local<<<8192, 32, 0, stream>>>(new_xyz, gp,
                                   Wq_h, Wkv_h,
                                   Wm0_h, bm0, Wm1_h, bm1, Wm2_h, bm2,
                                   Wl0_h, bl0, Wl1_h, bl1, Wl2_h, bl2,
                                   Wsk, Wwn, bwn,
                                   (float*)d_out, new_feat, skip, lp2v);

  k_gemm<<<128, 128, 0, stream>>>(new_feat, Wnlq_h, nlq_h, 8192, 64, 64);
  k_nlkv<<<512, 128, 0, stream>>>(points, Wnlkv_h, nlk_h, nlv_h);
  k_flash<<<512, 32, 0, stream>>>(nlq_h, nlk_h, nlv_h, Wnl_h, nlbuf);

  k_loc<<<128, 128, 0, stream>>>(lp2v, WpcR, bpc, locbuf);
  k_final<<<128, 128, 0, stream>>>(locbuf, nlbuf, skip, Wff_h,
                                   (float*)d_out + (size_t)8 * 3 * 1024);
}
